// BiLSTM_57440892617018
// MI455X (gfx1250) — matrix operand order repaired and checked by reading, not compiled, not run on hardware
//
#include <hip/hip_runtime.h>

// ---------------------------------------------------------------------------
// BiLSTM on MI455X (gfx1250): fused per-layer LSTM with W,U resident in the
// 320KB WGP LDS, bf16 WMMA (v_wmma_f32_16x16x32_bf16) for x@W and h@U,
// f32 accumulation, gate math in VALU, c-state in VGPRs, h via LDS.
// v3: async global->LDS x staging via inline asm (global_load_async_to_lds_b128
// + s_wait_asynccnt), zero-init accumulators (bias folded into gate epilogue),
// per-chunk operand preloading into distinct regs to overlap ds loads w/ WMMA.
// ---------------------------------------------------------------------------

typedef __bf16 bf16_t;
typedef __attribute__((ext_vector_type(8)))  __bf16 v8bf;
typedef __attribute__((ext_vector_type(16))) __bf16 v16bf;
typedef __attribute__((ext_vector_type(8)))  float  v8f;

#define T_STEPS  200
#define B_ROWS   2048
#define UNITS_K  128     // D == UNITS == 128 -> both layers share shapes
#define RPB      32      // batch rows per block
#define HPAD     136     // padded row stride (bf16 elems) for h / x tiles in LDS
#define WPAD     520     // padded row stride (bf16 elems) for W / U in LDS
#define NTHREADS 512     // 16 waves (wave32)

__device__ __forceinline__ float bf2f(bf16_t b) {
  unsigned short h = __builtin_bit_cast(unsigned short, b);
  unsigned u = ((unsigned)h) << 16;
  return __builtin_bit_cast(float, u);
}
__device__ __forceinline__ bf16_t f2bf(float f) {
  unsigned u = __builtin_bit_cast(unsigned, f);
  u += 0x7FFFu + ((u >> 16) & 1u);   // round-to-nearest-even
  unsigned short h = (unsigned short)(u >> 16);
  return __builtin_bit_cast(bf16_t, h);
}
__device__ __forceinline__ float sigm(float x) {
  // 1/(1+2^(-x*log2e)) : v_exp_f32 + v_rcp_f32
  return __builtin_amdgcn_rcpf(1.0f + __builtin_amdgcn_exp2f(-1.4426950408889634f * x));
}
__device__ __forceinline__ float tanh_f(float x) {
  // tanh(x) = 2*sigmoid(2x) - 1
  return 2.0f * __builtin_amdgcn_rcpf(1.0f + __builtin_amdgcn_exp2f(-2.8853900817779268f * x)) - 1.0f;
}
__device__ __forceinline__ v16bf cat16(v8bf lo, v8bf hi) {
  return __builtin_shufflevector(lo, hi, 0,1,2,3,4,5,6,7,8,9,10,11,12,13,14,15);
}
__device__ __forceinline__ v8f wmma_bf16(v16bf a, v16bf b, v8f c) {
  return __builtin_amdgcn_wmma_f32_16x16x32_bf16(false, a, false, b, (short)0, c, false, false);
}

// Async copy of 16 bytes global -> LDS (VFLAT async op, tracked by ASYNCcnt).
// LDS destination operand is the 32-bit LDS byte address (low 32 bits of the
// generic pointer into __shared__ memory).
__device__ __forceinline__ void async_copy_b128(const void* gsrc, void* lds_dst) {
  asm volatile("global_load_async_to_lds_b128 %0, %1, off"
               :
               : "v"((unsigned)(unsigned long long)lds_dst), "v"(gsrc)
               : "memory");
}
__device__ __forceinline__ void wait_asynccnt0() {
  asm volatile("s_wait_asynccnt 0x0" ::: "memory");
}

// ---------------------------------------------------------------------------
__global__ void f32_to_bf16_kernel(const float* __restrict__ src,
                                   bf16_t* __restrict__ dst, int n) {
  for (int i = blockIdx.x * blockDim.x + threadIdx.x; i < n;
       i += gridDim.x * blockDim.x)
    dst[i] = f2bf(src[i]);
}

__global__ void merge_kernel(const bf16_t* __restrict__ a,
                             const bf16_t* __restrict__ b,
                             float* __restrict__ out, int n) {
  for (int i = blockIdx.x * blockDim.x + threadIdx.x; i < n;
       i += gridDim.x * blockDim.x)
    out[i] = 0.5f * (bf2f(a[i]) + bf2f(b[i]));
}

// ---------------------------------------------------------------------------
// One LSTM layer, both directions in one launch (blockIdx.y: 0=fw, 1=bw).
// x layout: [B, T, 128] bf16.  W/U layout: [128, 512] bf16 (gates i,f,g,o).
// Each block: 32 batch rows; 16 waves; wave (rg,ug) computes the 16x16 tiles
// of gates i/f/g/o for rows rg*16..rg*16+15 and units ug*16..ug*16+15.
__global__ __launch_bounds__(NTHREADS)
void lstm_layer_kernel(const bf16_t* __restrict__ xfw, const bf16_t* __restrict__ xbw,
                       const bf16_t* __restrict__ Wfw, const bf16_t* __restrict__ Ufw,
                       const float*  __restrict__ bfw,
                       const bf16_t* __restrict__ Wbw, const bf16_t* __restrict__ Ubw,
                       const float*  __restrict__ bbw,
                       bf16_t* __restrict__ ofw, bf16_t* __restrict__ obw)
{
  const bool bw = (blockIdx.y != 0);
  const bf16_t* __restrict__ x  = bw ? xbw : xfw;
  const bf16_t* __restrict__ Wg = bw ? Wbw : Wfw;
  const bf16_t* __restrict__ Ug = bw ? Ubw : Ufw;
  const float*  __restrict__ bg = bw ? bbw : bfw;
  bf16_t* __restrict__ out = bw ? obw : ofw;

  const int b0   = blockIdx.x * RPB;
  const int tid  = threadIdx.x;
  const int lane = tid & 31;
  const int wave = tid >> 5;
  const int rg   = wave >> 3;          // row group 0..1
  const int n0   = (wave & 7) << 4;    // unit column base 0..112
  const int rowA = rg * 16 + (lane & 15);   // A-operand row for this lane
  const int koff = (lane >> 4) << 3;        // A-operand K sub-offset (0 or 8)
  const int nl   = n0 + (lane & 15);        // this lane's unit column

  extern __shared__ __align__(16) char smem_raw[];
  bf16_t* Wl = (bf16_t*)smem_raw;            // [128][WPAD]  ~130KB
  bf16_t* Ul = Wl + 128 * WPAD;              // [128][WPAD]  ~130KB
  bf16_t* hl = Ul + 128 * WPAD;              // [RPB][HPAD]  h state (bf16)
  bf16_t* xl = hl + RPB * HPAD;              // [RPB][HPAD]  staged x_t tile

  // Stage W and U into padded LDS images (resident for all 200 steps).
  for (int i = tid; i < 128 * 64; i += NTHREADS) {
    const int r = i >> 6, s = (i & 63) << 3;
    *(v8bf*)(Wl + r * WPAD + s) = *(const v8bf*)(Wg + r * 512 + s);
    *(v8bf*)(Ul + r * WPAD + s) = *(const v8bf*)(Ug + r * 512 + s);
  }
  // h(0) = 0
  for (int i = tid; i < RPB * HPAD; i += NTHREADS) hl[i] = f2bf(0.0f);

  // Per-lane biases for the four gates of this lane's unit column.
  const float bi = bg[nl], bff = bg[128 + nl], bgg = bg[256 + nl], bo = bg[384 + nl];

  v8f cst;                       // c-state: 16x16 tile, C/D layout, in VGPRs
#pragma unroll
  for (int r = 0; r < 8; ++r) cst[r] = 0.0f;

  const int xrow = tid >> 4;               // staging: row 0..31
  const int xseg = (tid & 15) << 3;        // staging: col 0,8,..,120

  for (int s = 0; s < T_STEPS; ++s) {
    const int t = bw ? (T_STEPS - 1 - s) : s;

    // Stage this timestep's x tile (32x128 bf16 = 8KB) into LDS, async.
    const bf16_t* xsrc = x + ((size_t)(b0 + xrow) * T_STEPS + t) * UNITS_K + xseg;
    bf16_t* xdst = xl + xrow * HPAD + xseg;
    async_copy_b128(xsrc, xdst);
    if (s + 1 < T_STEPS) {
      const int tn = bw ? (t - 1) : (t + 1);
      __builtin_prefetch(x + ((size_t)(b0 + xrow) * T_STEPS + tn) * UNITS_K + xseg, 0, 0);
    }
    wait_asynccnt0();
    __syncthreads();   // x staged; previous step's h writes visible

    // Accumulators start at zero; bias is folded into the gate epilogue.
    v8f acc[4];
#pragma unroll
    for (int g = 0; g < 4; ++g)
#pragma unroll
      for (int r = 0; r < 8; ++r) acc[g][r] = 0.0f;

    // z = x_t @ W + h @ U  over K=128 in 4 chunks of 32.  All operands of a
    // chunk are preloaded into distinct registers so the ds_loads drain under
    // the WMMAs instead of serializing on register reuse.
#pragma unroll
    for (int kk = 0; kk < UNITS_K; kk += 32) {
      const v16bf a_x = cat16(*(const v8bf*)(xl + rowA * HPAD + kk + koff),
                              *(const v8bf*)(xl + rowA * HPAD + kk + koff + 16));
      const v16bf a_h = cat16(*(const v8bf*)(hl + rowA * HPAD + kk + koff),
                              *(const v8bf*)(hl + rowA * HPAD + kk + koff + 16));
      v16bf bW[4], bU[4];
#pragma unroll
      for (int g = 0; g < 4; ++g) {
        const int cb = g * 128 + n0 + (lane & 15);
#pragma unroll
        for (int j = 0; j < 16; ++j) {
          const int krow = kk + koff + (j < 8 ? j : j + 8);
          bW[g][j] = Wl[krow * WPAD + cb];
          bU[g][j] = Ul[krow * WPAD + cb];
        }
      }
#pragma unroll
      for (int g = 0; g < 4; ++g) acc[g] = wmma_bf16(a_x, bW[g], acc[g]);
#pragma unroll
      for (int g = 0; g < 4; ++g) acc[g] = wmma_bf16(a_h, bU[g], acc[g]);
    }

    // Gate math: c = sig(f+bf)*c + sig(i+bi)*tanh(g+bg); h = sig(o+bo)*tanh(c)
    v8f hnew;
#pragma unroll
    for (int r = 0; r < 8; ++r) {
      const float cv = sigm(acc[1][r] + bff) * cst[r] +
                       sigm(acc[0][r] + bi) * tanh_f(acc[2][r] + bgg);
      cst[r] = cv;
      hnew[r] = sigm(acc[3][r] + bo) * tanh_f(cv);
    }

    __syncthreads();   // all waves done reading hl before we overwrite it

    // C/D layout: VGPR r holds row (r | r+8) for lanes (0-15 | 16-31).
    const int mbase = rg * 16 + ((lane >> 4) << 3);
#pragma unroll
    for (int r = 0; r < 8; ++r) {
      const int m = mbase + r;
      const bf16_t hb = f2bf(hnew[r]);
      hl[m * HPAD + nl] = hb;                                      // next step's A
      out[((size_t)(b0 + m) * T_STEPS + t) * UNITS_K + nl] = hb;   // sequence output
    }
  }
}

// ---------------------------------------------------------------------------
extern "C" void kernel_launch(void* const* d_in, const int* in_sizes, int n_in,
                              void* d_out, int out_size, void* d_ws, size_t ws_size,
                              hipStream_t stream) {
  (void)in_sizes; (void)n_in; (void)out_size; (void)ws_size;
  const float* x   = (const float*)d_in[0];
  const float* Wf  = (const float*)d_in[1];
  const float* Uf  = (const float*)d_in[2];
  const float* bf_ = (const float*)d_in[3];
  const float* Wb  = (const float*)d_in[4];
  const float* Ub  = (const float*)d_in[5];
  const float* bb_ = (const float*)d_in[6];
  float* out = (float*)d_out;

  const size_t S = (size_t)B_ROWS * T_STEPS * UNITS_K;   // 52,428,800 elems
  bf16_t* ws  = (bf16_t*)d_ws;
  bf16_t* xbf = ws;              // x in bf16; reused as layer-1 fw output
  bf16_t* h0f = ws + S;
  bf16_t* h0b = ws + 2 * S;
  bf16_t* h1b = ws + 3 * S;
  bf16_t* wts = ws + 4 * S;      // bf16 weights: Wf,Uf,Wb,Ub each [2][128][512]
  const size_t WSZ = (size_t)2 * 128 * 512;
  bf16_t* Wfc = wts;
  bf16_t* Ufc = wts + WSZ;
  bf16_t* Wbc = wts + 2 * WSZ;
  bf16_t* Ubc = wts + 3 * WSZ;
  const int LOFF = 128 * 512;    // per-layer weight offset

  f32_to_bf16_kernel<<<8192, 256, 0, stream>>>(x,  xbf, (int)S);
  f32_to_bf16_kernel<<<256,  256, 0, stream>>>(Wf, Wfc, (int)WSZ);
  f32_to_bf16_kernel<<<256,  256, 0, stream>>>(Uf, Ufc, (int)WSZ);
  f32_to_bf16_kernel<<<256,  256, 0, stream>>>(Wb, Wbc, (int)WSZ);
  f32_to_bf16_kernel<<<256,  256, 0, stream>>>(Ub, Ubc, (int)WSZ);

  const size_t SH = ((size_t)2 * 128 * WPAD + (size_t)2 * RPB * HPAD) * sizeof(bf16_t);
  dim3 grid(B_ROWS / RPB, 2);   // 64 batch tiles x {fw,bw}

  // Layer 0: both directions read x; bw runs its scan time-reversed internally.
  lstm_layer_kernel<<<grid, NTHREADS, SH, stream>>>(
      xbf, xbf, Wfc, Ufc, bf_, Wbc, Ubc, bb_, h0f, h0b);

  // Layer 1: fw consumes h0f, bw consumes h0b; fw output reuses x's buffer.
  lstm_layer_kernel<<<grid, NTHREADS, SH, stream>>>(
      h0f, h0b, Wfc + LOFF, Ufc + LOFF, bf_ + 512,
      Wbc + LOFF, Ubc + LOFF, bb_ + 512, xbf, h1b);

  // merge_mode 'ave'
  merge_kernel<<<8192, 256, 0, stream>>>(xbf, h1b, out, (int)S);
}
